// Model_2353642078537
// MI455X (gfx1250) — compile-verified, matrix operand order checked by reading, not run on hardware
//
#include <hip/hip_runtime.h>
#include <cstdint>

#define BATCH 32
#define NPTS  16384
#define NP    64        // NPOINT
#define SMP   256       // SAMPLE
#define DMODEL 256
#define TOK   65        // 1 cls + 64
#define EPSV  1e-5f

typedef __attribute__((ext_vector_type(16))) _Float16 v16h;
typedef __attribute__((ext_vector_type(8)))  _Float16 v8h;
typedef __attribute__((ext_vector_type(8)))  float    v8f;

union V16H { v16h v; v8h h[2]; };

__device__ __forceinline__ float activate(float v, int act) {
  if (act == 1) return fmaxf(v, 0.f);
  if (act == 2) return 0.5f * v * (1.f + erff(v * 0.70710678118f)); // exact gelu
  if (act == 3) return v > 0.f ? v : 0.2f * v;                     // leaky 0.2
  return v;
}

// CDNA5 async global->LDS copy (128b per lane), tracked by ASYNCcnt.
// lds_addr: wave-relative LDS byte address (low 32 bits of generic pointer).
__device__ __forceinline__ void async_copy_b128(unsigned lds_addr, const void* gaddr) {
  asm volatile("global_load_async_to_lds_b128 %0, %1, off"
               :: "v"(lds_addr), "v"(gaddr) : "memory");
}
__device__ __forceinline__ void async_wait0() {
  asm volatile("s_wait_asynccnt 0x0" ::: "memory");
}

// ---------------- small utility kernels ----------------

__global__ void affine_kernel(const float* g, const float* b, const float* m,
                              const float* v, float* s, float* sh, int n) {
  int i = blockIdx.x * 256 + threadIdx.x;
  if (i >= n) return;
  float sc = g[i] * rsqrtf(v[i] + EPSV);
  s[i] = sc;
  sh[i] = b[i] - m[i] * sc;
}

__global__ void convert_kernel(const float* s, _Float16* d, int n) {
  int i = blockIdx.x * 256 + threadIdx.x;
  if (i < n) d[i] = (_Float16)s[i];
}

// local_w is (256,128); extract W1 = local_w[:, :64] as f16 row-major (256x64)
__global__ void w1conv_kernel(const float* lw, _Float16* w1) {
  int i = blockIdx.x * 256 + threadIdx.x;
  if (i >= 256 * 64) return;
  w1[i] = (_Float16)lw[(i >> 6) * 128 + (i & 63)];
}

__global__ void cls_kernel(const float* cls, float* t) {
  // cls_token (1,D,1) -> t[b,0,d] = cls[d]
  t[(long)blockIdx.x * TOK * DMODEL + threadIdx.x] = cls[threadIdx.x];
}

// ---------------- fused input MLP: 3 -> 64 -> 64 (bn + leaky) ----------------

__global__ __launch_bounds__(256) void embed_kernel(
    const float* __restrict__ x, const float* __restrict__ c1w,
    const float* __restrict__ a1s, const float* __restrict__ a1h,
    const float* __restrict__ c2w, const float* __restrict__ a2s,
    const float* __restrict__ a2h, float* __restrict__ xyz,
    _Float16* __restrict__ pts) {
  __shared__ float w2[64 * 64];
  __shared__ float w1[64 * 3];
  __shared__ float s1[64], h1[64], s2[64], h2[64];
  int tid = threadIdx.x;
  for (int j = tid; j < 4096; j += 256) w2[j] = c2w[j];
  if (tid < 192) w1[tid] = c1w[tid];
  if (tid < 64) { s1[tid] = a1s[tid]; h1[tid] = a1h[tid]; s2[tid] = a2s[tid]; h2[tid] = a2h[tid]; }
  __syncthreads();

  long p = (long)blockIdx.x * 256 + tid;   // global point id
  int b = (int)(p >> 14);
  int n = (int)(p & (NPTS - 1));
  float x0 = x[((long)b * 3 + 0) * NPTS + n];
  float x1 = x[((long)b * 3 + 1) * NPTS + n];
  float x2 = x[((long)b * 3 + 2) * NPTS + n];
  xyz[p * 3 + 0] = x0; xyz[p * 3 + 1] = x1; xyz[p * 3 + 2] = x2;

  float h[64];
  #pragma unroll
  for (int c = 0; c < 64; ++c) {
    float v = w1[c * 3] * x0 + w1[c * 3 + 1] * x1 + w1[c * 3 + 2] * x2;
    v = v * s1[c] + h1[c];
    h[c] = v > 0.f ? v : 0.2f * v;
  }
  for (int e = 0; e < 64; ++e) {
    float v = 0.f;
    #pragma unroll
    for (int c = 0; c < 64; ++c) v += w2[e * 64 + c] * h[c];
    v = v * s2[e] + h2[e];
    v = v > 0.f ? v : 0.2f * v;
    pts[p * 64 + e] = (_Float16)v;
  }
}

// ---------------- FPS: one workgroup per batch, dist resident in LDS ----------------

__global__ __launch_bounds__(1024) void fps_kernel(const float* __restrict__ xyz,
                                                   int* __restrict__ fps_idx,
                                                   float* __restrict__ new_xyz) {
  extern __shared__ float fps_dist[];     // NPTS floats (64 KB)
  __shared__ float rv[32];
  __shared__ int   ri[32];
  __shared__ int   sfar;
  int b = blockIdx.x, tid = threadIdx.x;
  for (int i = tid; i < NPTS; i += 1024) fps_dist[i] = 1e10f;
  int far = 0;
  __syncthreads();

  for (int it = 0; it < NP; ++it) {
    float cx = xyz[((long)b * NPTS + far) * 3 + 0];
    float cy = xyz[((long)b * NPTS + far) * 3 + 1];
    float cz = xyz[((long)b * NPTS + far) * 3 + 2];
    if (tid == 0) {
      fps_idx[b * NP + it] = far;
      new_xyz[(b * NP + it) * 3 + 0] = cx;
      new_xyz[(b * NP + it) * 3 + 1] = cy;
      new_xyz[(b * NP + it) * 3 + 2] = cz;
    }
    float bv = -1.f; int bi = 0;
    for (int i = tid; i < NPTS; i += 1024) {
      float dx = xyz[((long)b * NPTS + i) * 3 + 0] - cx;
      float dy = xyz[((long)b * NPTS + i) * 3 + 1] - cy;
      float dz = xyz[((long)b * NPTS + i) * 3 + 2] - cz;
      float d = dx * dx + dy * dy + dz * dz;
      float dd = fminf(fps_dist[i], d);
      fps_dist[i] = dd;
      if (dd > bv || (dd == bv && i < bi)) { bv = dd; bi = i; }
    }
    #pragma unroll
    for (int o = 16; o; o >>= 1) {
      float ov = __shfl_down(bv, o, 32);
      int   oi = __shfl_down(bi, o, 32);
      if (ov > bv || (ov == bv && oi < bi)) { bv = ov; bi = oi; }
    }
    if ((tid & 31) == 0) { rv[tid >> 5] = bv; ri[tid >> 5] = bi; }
    __syncthreads();
    if (tid < 32) {
      float bv2 = rv[tid]; int bi2 = ri[tid];
      #pragma unroll
      for (int o = 16; o; o >>= 1) {
        float ov = __shfl_down(bv2, o, 32);
        int   oi = __shfl_down(bi2, o, 32);
        if (ov > bv2 || (ov == bv2 && oi < bi2)) { bv2 = ov; bi2 = oi; }
      }
      if (tid == 0) sfar = bi2;
    }
    __syncthreads();
    far = sfar;
  }
}

// ---------------- kNN: 256 smallest of 16384, LDS radix select ----------------

__global__ __launch_bounds__(256) void knn_kernel(const float* __restrict__ xyz,
                                                  const float* __restrict__ new_xyz,
                                                  int* __restrict__ knn) {
  extern __shared__ unsigned int knn_keys[];   // NPTS uints (64 KB)
  __shared__ int hist[256];
  __shared__ unsigned int s_prefix;
  __shared__ int s_want, s_cnt;
  int g = blockIdx.x;          // b*64 + s
  int b = g >> 6;
  int tid = threadIdx.x;
  float cx = new_xyz[g * 3 + 0], cy = new_xyz[g * 3 + 1], cz = new_xyz[g * 3 + 2];

  for (int i = tid; i < NPTS; i += 256) {
    float dx = xyz[((long)b * NPTS + i) * 3 + 0] - cx;
    float dy = xyz[((long)b * NPTS + i) * 3 + 1] - cy;
    float dz = xyz[((long)b * NPTS + i) * 3 + 2] - cz;
    knn_keys[i] = __float_as_uint(dx * dx + dy * dy + dz * dz);  // >=0: bit order == value order
  }
  __syncthreads();

  unsigned int prefix = 0; int want = SMP;
  for (int pb = 3; pb >= 0; --pb) {
    hist[tid] = 0;
    __syncthreads();
    unsigned int maskAbove = (pb == 3) ? 0u : (0xFFFFFFFFu << (8 * (pb + 1)));
    for (int i = tid; i < NPTS; i += 256) {
      unsigned int k = knn_keys[i];
      if ((k & maskAbove) == prefix) atomicAdd(&hist[(k >> (8 * pb)) & 255], 1);
    }
    __syncthreads();
    if (tid == 0) {
      int cum = 0;
      for (int bin = 0; bin < 256; ++bin) {
        int c = hist[bin];
        if (cum + c >= want) { s_prefix = prefix | ((unsigned int)bin << (8 * pb)); s_want = want - cum; break; }
        cum += c;
      }
    }
    __syncthreads();
    prefix = s_prefix; want = s_want;
    __syncthreads();
  }
  unsigned int pivot = prefix;
  if (tid == 0) s_cnt = 0;
  __syncthreads();
  for (int i = tid; i < NPTS; i += 256)
    if (knn_keys[i] < pivot) { int pos = atomicAdd(&s_cnt, 1); knn[(long)g * SMP + pos] = i; }
  __syncthreads();
  for (int i = tid; i < NPTS; i += 256)
    if (knn_keys[i] == pivot) { int pos = atomicAdd(&s_cnt, 1); if (pos < SMP) knn[(long)g * SMP + pos] = i; }
}

// ---------------- per-group bias: (center @ (W2-W1)^T)*scale + shift ----------------

__global__ __launch_bounds__(256) void cbias_kernel(const _Float16* __restrict__ pts,
                                                    const int* __restrict__ fps_idx,
                                                    const float* __restrict__ local_w,
                                                    const float* __restrict__ lscale,
                                                    const float* __restrict__ lshift,
                                                    float* __restrict__ cbias) {
  int g = blockIdx.x, b = g >> 6, s = g & 63, e = threadIdx.x;
  int ci = fps_idx[b * NP + s];
  const _Float16* c = pts + ((long)b * NPTS + ci) * 64;
  float acc = 0.f;
  #pragma unroll
  for (int d = 0; d < 64; ++d) {
    float cd = (float)c[d];
    acc += cd * (local_w[e * 128 + 64 + d] - local_w[e * 128 + d]);
  }
  cbias[(long)g * 256 + e] = acc * lscale[e] + lshift[e];
}

// ---------------- grouped local conv: async-LDS staged WMMA + fused relu + max over k --------

__global__ __launch_bounds__(256) void localconv_kernel(
    const _Float16* __restrict__ pts, const int* __restrict__ knn,
    const _Float16* __restrict__ W1h, const float* __restrict__ lscale,
    const float* __restrict__ cbias, float* __restrict__ t) {
  extern __shared__ char lc_sm[];
  _Float16* Ash = (_Float16*)lc_sm;            // 256 x 64 gathered samples (f16)
  _Float16* Wsh = Ash + 256 * 64;              // 256 x 64 weights (f16)
  int*   tokmax = (int*)(Wsh + 256 * 64);      // 256 (float bits, vals >= 0)
  float* cbs    = (float*)(tokmax + 256);
  float* scs    = cbs + 256;
  int g = blockIdx.x, b = g >> 6, s = g & 63;
  int tid = threadIdx.x;

  { // gather one sample per thread (128B each) via async global->LDS
    int idx = knn[(long)g * SMP + tid];
    const char* src = (const char*)(pts + ((long)b * NPTS + idx) * 64);
    unsigned dst = (unsigned)(uintptr_t)(Ash + tid * 64);
    #pragma unroll
    for (int j = 0; j < 8; ++j) async_copy_b128(dst + j * 16, src + j * 16);
  }
  { // stage W1 (32KB) via async global->LDS
    const char* src = (const char*)W1h;
    unsigned dst = (unsigned)(uintptr_t)Wsh;
    #pragma unroll
    for (int j = tid; j < 2048; j += 256) async_copy_b128(dst + j * 16, src + j * 16);
  }
  tokmax[tid] = 0;                    // relu output >= 0, int-bit max is valid
  cbs[tid] = cbias[(long)g * 256 + tid];
  scs[tid] = lscale[tid];
  async_wait0();                      // own wave's ASYNCcnt == 0
  __syncthreads();                    // all waves' LDS writes visible

  int wave = tid >> 5, lane = tid & 31;
  int lrow = lane & 15, off = (lane >> 4) * 8;
  for (int mt = wave * 2; mt < wave * 2 + 2; ++mt) {
    V16H a0, a1;
    const _Float16* ap = Ash + (mt * 16 + lrow) * 64;
    a0.h[0] = *(const v8h*)(ap + off);       a0.h[1] = *(const v8h*)(ap + off + 16);
    a1.h[0] = *(const v8h*)(ap + 32 + off);  a1.h[1] = *(const v8h*)(ap + 32 + off + 16);
    for (int nt = 0; nt < 16; ++nt) {
      const _Float16* wp = Wsh + (nt * 16 + lrow) * 64;
      V16H b0, b1;
      b0.h[0] = *(const v8h*)(wp + off);       b0.h[1] = *(const v8h*)(wp + off + 16);
      b1.h[0] = *(const v8h*)(wp + 32 + off);  b1.h[1] = *(const v8h*)(wp + 32 + off + 16);
      v8f acc = {};
      acc = __builtin_amdgcn_wmma_f32_16x16x32_f16(false, a0.v, false, b0.v, (short)0, acc, false, false);
      acc = __builtin_amdgcn_wmma_f32_16x16x32_f16(false, a1.v, false, b1.v, (short)0, acc, false, false);
      int e = nt * 16 + lrow;
      float sc = scs[e], cb = cbs[e];
      float mx = 0.f;
      #pragma unroll
      for (int r = 0; r < 8; ++r) {
        float v = fmaxf(acc[r] * sc + cb, 0.f);
        mx = fmaxf(mx, v);
      }
      float o = __shfl_xor(mx, 16, 32);
      mx = fmaxf(mx, o);
      if (lane < 16) atomicMax(&tokmax[e], __float_as_int(mx));
    }
  }
  __syncthreads();
  t[((long)b * TOK + (s + 1)) * DMODEL + tid] = __int_as_float(tokmax[tid]);
}

// ---------------- layernorm (writes f16 activations) ----------------

__global__ __launch_bounds__(256) void ln_kernel(const float* __restrict__ t,
                                                 const float* __restrict__ g,
                                                 const float* __restrict__ bt,
                                                 _Float16* __restrict__ y) {
  __shared__ float red[8];
  __shared__ float bc[2];
  long row = blockIdx.x;   // b*65 + r
  int tid = threadIdx.x;
  float v = t[row * DMODEL + tid];
  float s = v;
  #pragma unroll
  for (int o = 16; o; o >>= 1) s += __shfl_down(s, o, 32);
  if ((tid & 31) == 0) red[tid >> 5] = s;
  __syncthreads();
  if (tid == 0) { float a = 0; for (int i = 0; i < 8; ++i) a += red[i]; bc[0] = a * (1.f / 256.f); }
  __syncthreads();
  float mu = bc[0];
  float d = v - mu;
  float q = d * d;
  #pragma unroll
  for (int o = 16; o; o >>= 1) q += __shfl_down(q, o, 32);
  if ((tid & 31) == 0) red[tid >> 5] = q;
  __syncthreads();
  if (tid == 0) { float a = 0; for (int i = 0; i < 8; ++i) a += red[i]; bc[1] = a * (1.f / 256.f); }
  __syncthreads();
  float rs = rsqrtf(bc[1] + EPSV);
  y[row * DMODEL + tid] = (_Float16)(d * rs * g[tid] + bt[tid]);
}

// ---------------- generic WMMA GEMM: out = act(A@W^T * scale + bias + residual) ----------------
// A: f16 [batch][M][K] row-major; W: f16 [N][K] row-major (== weight [out,in])

__global__ __launch_bounds__(32) void gemm_wmma_kernel(
    const _Float16* __restrict__ A, long aStride, const _Float16* __restrict__ W,
    const float* scale, const float* bias, const float* residual, long resStride,
    float* outF, _Float16* outH, long outStride, int M, int N, int K, int act) {
  int bt = blockIdx.z;
  int n0 = blockIdx.x * 16, m0 = blockIdx.y * 16;
  int lane = threadIdx.x;
  int ar = m0 + (lane & 15); ar = ar < M ? ar : M - 1;
  int wr = n0 + (lane & 15); wr = wr < N ? wr : N - 1;
  int off = (lane >> 4) * 8;
  const _Float16* ap = A + (long)bt * aStride + (long)ar * K;
  const _Float16* wp = W + (long)wr * K;
  v8f acc = {};
  for (int k0 = 0; k0 < K; k0 += 32) {
    // pull next k-tiles toward the WGP while this WMMA runs (speculative, drop-safe)
    __builtin_prefetch(ap + k0 + 32, 0, 3);
    __builtin_prefetch(wp + k0 + 32, 0, 3);
    V16H a, bm;
    a.h[0]  = *(const v8h*)(ap + k0 + off);
    a.h[1]  = *(const v8h*)(ap + k0 + off + 16);
    bm.h[0] = *(const v8h*)(wp + k0 + off);
    bm.h[1] = *(const v8h*)(wp + k0 + off + 16);
    acc = __builtin_amdgcn_wmma_f32_16x16x32_f16(false, a.v, false, bm.v, (short)0, acc, false, false);
  }
  int n = n0 + (lane & 15);
  float sc = 1.f, bs = 0.f;
  if (n < N) { if (scale) sc = scale[n]; if (bias) bs = bias[n]; }
  #pragma unroll
  for (int r = 0; r < 8; ++r) {
    int m = m0 + r + 8 * (lane >> 4);
    if (m >= M || n >= N) continue;
    float v = acc[r] * sc + bs;
    if (residual) v += residual[(long)bt * resStride + (long)m * N + n];
    v = activate(v, act);
    long o = (long)bt * outStride + (long)m * N + n;
    if (outF) outF[o] = v;
    if (outH) outH[o] = (_Float16)v;
  }
}

// ---------------- attention: one block per (batch, head), everything in LDS ----------------

__global__ __launch_bounds__(96) void attn_kernel(const float* __restrict__ qkv,
                                                  _Float16* __restrict__ o16) {
  extern __shared__ float at_sm[];
  float* q = at_sm;            // 65*64
  float* k = q + TOK * 64;     // 65*64
  float* v = k + TOK * 64;     // 65*64
  float* p = v + TOK * 64;     // 65*65
  int b = blockIdx.x >> 2, h = blockIdx.x & 3;
  int tid = threadIdx.x;
  for (int i = tid; i < TOK * 64; i += 96) {
    int r = i >> 6, c = i & 63;
    const float* base = qkv + ((long)(b * TOK + r)) * 768 + h * 64 + c;
    q[i] = base[0]; k[i] = base[256]; v[i] = base[512];
  }
  __syncthreads();
  if (tid < TOK) {
    float mx = -1e30f;
    for (int j = 0; j < TOK; ++j) {
      float acc = 0.f;
      #pragma unroll 8
      for (int c = 0; c < 64; ++c) acc += q[tid * 64 + c] * k[j * 64 + c];
      acc *= 0.125f;   // hd^-0.5
      p[tid * TOK + j] = acc;
      mx = fmaxf(mx, acc);
    }
    float s = 0.f;
    for (int j = 0; j < TOK; ++j) { float e = expf(p[tid * TOK + j] - mx); p[tid * TOK + j] = e; s += e; }
    float inv = 1.f / s;
    for (int c = 0; c < 64; ++c) {
      float acc = 0.f;
      for (int j = 0; j < TOK; ++j) acc += p[tid * TOK + j] * v[j * 64 + c];
      o16[((long)(b * TOK + tid)) * DMODEL + h * 64 + c] = (_Float16)(acc * inv);
    }
  }
}

// ---------------- pooling: max || mean over 65 tokens, write f16 ----------------

__global__ void pool_kernel(const float* __restrict__ h, _Float16* __restrict__ pooled) {
  int i = blockIdx.x * 256 + threadIdx.x;   // B*1024
  int b = i >> 10, e = i & 1023;
  const float* p = h + (long)b * TOK * 1024 + e;
  float mx = -1e30f, sm = 0.f;
  for (int r = 0; r < TOK; ++r) { float v = p[(long)r * 1024]; mx = fmaxf(mx, v); sm += v; }
  pooled[b * 2048 + e] = (_Float16)mx;
  pooled[b * 2048 + 1024 + e] = (_Float16)(sm * (1.f / 65.f));
}

__global__ void sescale_kernel(const _Float16* z1, const float* s2, _Float16* z) {
  int i = blockIdx.x * 256 + threadIdx.x;   // B*512
  float sv = 1.f / (1.f + expf(-s2[i]));
  z[i] = (_Float16)((float)z1[i] * sv);
}

// ---------------- host ----------------

static inline dim3 gemm_grid(int M, int N, int batch) {
  return dim3((N + 15) / 16, (M + 15) / 16, batch);
}

extern "C" void kernel_launch(void* const* d_in, const int* in_sizes, int n_in,
                              void* d_out, int out_size, void* d_ws, size_t ws_size,
                              hipStream_t stream) {
  (void)in_sizes; (void)n_in; (void)out_size; (void)ws_size;
  const float* x        = (const float*)d_in[0];
  const float* conv1_w  = (const float*)d_in[1];
  const float* bn1_g = (const float*)d_in[2], *bn1_b = (const float*)d_in[3];
  const float* bn1_m = (const float*)d_in[4], *bn1_v = (const float*)d_in[5];
  const float* conv2_w  = (const float*)d_in[6];
  const float* bn2_g = (const float*)d_in[7], *bn2_b = (const float*)d_in[8];
  const float* bn2_m = (const float*)d_in[9], *bn2_v = (const float*)d_in[10];
  const float* local_w  = (const float*)d_in[11];
  const float* lbn_g = (const float*)d_in[12], *lbn_b = (const float*)d_in[13];
  const float* lbn_m = (const float*)d_in[14], *lbn_v = (const float*)d_in[15];
  const float* cls_tok  = (const float*)d_in[16];
  const float* ln1_g = (const float*)d_in[17], *ln1_b = (const float*)d_in[18];
  const float* qkv_w = (const float*)d_in[19];
  const float* proj_w = (const float*)d_in[20], *proj_b = (const float*)d_in[21];
  const float* ln2_g = (const float*)d_in[22], *ln2_b = (const float*)d_in[23];
  const float* fc1_w = (const float*)d_in[24], *fc1_b = (const float*)d_in[25];
  const float* fc2_w = (const float*)d_in[26], *fc2_b = (const float*)d_in[27];
  const float* norm_g = (const float*)d_in[28], *norm_b = (const float*)d_in[29];
  const float* h0_w = (const float*)d_in[30];
  const float* h0bn_g = (const float*)d_in[31], *h0bn_b = (const float*)d_in[32];
  const float* h0bn_m = (const float*)d_in[33], *h0bn_v = (const float*)d_in[34];
  const float* h1_w = (const float*)d_in[35];
  const float* h1bn_g = (const float*)d_in[36], *h1bn_b = (const float*)d_in[37];
  const float* h1bn_m = (const float*)d_in[38], *h1bn_v = (const float*)d_in[39];
  const float* se1_w = (const float*)d_in[40];
  const float* sebn_g = (const float*)d_in[41], *sebn_b = (const float*)d_in[42];
  const float* sebn_m = (const float*)d_in[43], *sebn_v = (const float*)d_in[44];
  const float* se2_w = (const float*)d_in[45];
  const float* h2_w = (const float*)d_in[46], *h2_b = (const float*)d_in[47];

  char* ws = (char*)d_ws;
  size_t off = 0;
  auto alloc = [&](size_t bytes) -> char* {
    off = (off + 255) & ~(size_t)255;
    char* p = ws + off;
    off += bytes;
    return p;
  };

  float*    xyz      = (float*)alloc((size_t)BATCH * NPTS * 3 * 4);
  _Float16* pts      = (_Float16*)alloc((size_t)BATCH * NPTS * 64 * 2);
  int*      fpsIdx   = (int*)alloc((size_t)BATCH * NP * 4);
  float*    newXyz   = (float*)alloc((size_t)BATCH * NP * 3 * 4);
  int*      knn      = (int*)alloc((size_t)BATCH * NP * SMP * 4);
  float* bn1s = (float*)alloc(64 * 4), *bn1h = (float*)alloc(64 * 4);
  float* bn2s = (float*)alloc(64 * 4), *bn2h = (float*)alloc(64 * 4);
  float* lbs  = (float*)alloc(256 * 4), *lbh = (float*)alloc(256 * 4);
  float* h0s  = (float*)alloc(1024 * 4), *h0h = (float*)alloc(1024 * 4);
  float* h1s  = (float*)alloc(512 * 4), *h1h = (float*)alloc(512 * 4);
  float* ses  = (float*)alloc(64 * 4), *seh = (float*)alloc(64 * 4);
  _Float16* W1h     = (_Float16*)alloc(256 * 64 * 2);
  float*    cbias   = (float*)alloc((size_t)BATCH * NP * 256 * 4);
  _Float16* qkvW16  = (_Float16*)alloc((size_t)2 * 768 * 256 * 2);
  _Float16* projW16 = (_Float16*)alloc((size_t)2 * 256 * 256 * 2);
  _Float16* fc1W16  = (_Float16*)alloc((size_t)2 * 512 * 256 * 2);
  _Float16* fc2W16  = (_Float16*)alloc((size_t)2 * 256 * 512 * 2);
  _Float16* h0W16   = (_Float16*)alloc((size_t)1024 * 256 * 2);
  _Float16* h1W16   = (_Float16*)alloc((size_t)512 * 2048 * 2);
  _Float16* se1W16  = (_Float16*)alloc((size_t)64 * 512 * 2);
  _Float16* se2W16  = (_Float16*)alloc((size_t)512 * 64 * 2);
  _Float16* h2W16   = (_Float16*)alloc((size_t)40 * 512 * 2);
  float*    t       = (float*)alloc((size_t)BATCH * TOK * DMODEL * 4);
  _Float16* y16     = (_Float16*)alloc((size_t)BATCH * TOK * DMODEL * 2);
  float*    qkvbuf  = (float*)alloc((size_t)BATCH * TOK * 768 * 4);
  _Float16* o16     = (_Float16*)alloc((size_t)BATCH * TOK * DMODEL * 2);
  _Float16* mlp16   = (_Float16*)alloc((size_t)BATCH * TOK * 512 * 2);
  float*    h0out   = (float*)alloc((size_t)BATCH * TOK * 1024 * 4);
  _Float16* pooled16= (_Float16*)alloc((size_t)BATCH * 2048 * 2);
  _Float16* z116    = (_Float16*)alloc((size_t)BATCH * 512 * 2);
  _Float16* s116    = (_Float16*)alloc((size_t)BATCH * 64 * 2);
  float*    s2buf   = (float*)alloc((size_t)BATCH * 512 * 4);
  _Float16* z16     = (_Float16*)alloc((size_t)BATCH * 512 * 2);

  (void)hipFuncSetAttribute((const void*)fps_kernel,       hipFuncAttributeMaxDynamicSharedMemorySize, 65536);
  (void)hipFuncSetAttribute((const void*)knn_kernel,       hipFuncAttributeMaxDynamicSharedMemorySize, 65536);
  (void)hipFuncSetAttribute((const void*)localconv_kernel, hipFuncAttributeMaxDynamicSharedMemorySize, 70000);
  (void)hipFuncSetAttribute((const void*)attn_kernel,      hipFuncAttributeMaxDynamicSharedMemorySize, 67000);

  // BN affine precompute
  affine_kernel<<<1, 256, 0, stream>>>(bn1_g, bn1_b, bn1_m, bn1_v, bn1s, bn1h, 64);
  affine_kernel<<<1, 256, 0, stream>>>(bn2_g, bn2_b, bn2_m, bn2_v, bn2s, bn2h, 64);
  affine_kernel<<<1, 256, 0, stream>>>(lbn_g, lbn_b, lbn_m, lbn_v, lbs, lbh, 256);
  affine_kernel<<<4, 256, 0, stream>>>(h0bn_g, h0bn_b, h0bn_m, h0bn_v, h0s, h0h, 1024);
  affine_kernel<<<2, 256, 0, stream>>>(h1bn_g, h1bn_b, h1bn_m, h1bn_v, h1s, h1h, 512);
  affine_kernel<<<1, 256, 0, stream>>>(sebn_g, sebn_b, sebn_m, sebn_v, ses, seh, 64);

  // weight conversions to f16
  auto conv = [&](const float* s, _Float16* d, int n) {
    convert_kernel<<<(n + 255) / 256, 256, 0, stream>>>(s, d, n);
  };
  conv(qkv_w,  qkvW16,  2 * 768 * 256);
  conv(proj_w, projW16, 2 * 256 * 256);
  conv(fc1_w,  fc1W16,  2 * 512 * 256);
  conv(fc2_w,  fc2W16,  2 * 256 * 512);
  conv(h0_w,   h0W16,   1024 * 256);
  conv(h1_w,   h1W16,   512 * 2048);
  conv(se1_w,  se1W16,  64 * 512);
  conv(se2_w,  se2W16,  512 * 64);
  conv(h2_w,   h2W16,   40 * 512);
  w1conv_kernel<<<64, 256, 0, stream>>>(local_w, W1h);

  // point embedding (fused conv1+conv2), FPS, kNN, grouping bias
  embed_kernel<<<(BATCH * NPTS) / 256, 256, 0, stream>>>(x, conv1_w, bn1s, bn1h,
                                                         conv2_w, bn2s, bn2h, xyz, pts);
  fps_kernel<<<BATCH, 1024, NPTS * 4, stream>>>(xyz, fpsIdx, newXyz);
  knn_kernel<<<BATCH * NP, 256, NPTS * 4, stream>>>(xyz, newXyz, knn);
  cbias_kernel<<<BATCH * NP, 256, 0, stream>>>(pts, fpsIdx, local_w, lbs, lbh, cbias);
  localconv_kernel<<<BATCH * NP, 256, 256 * 64 * 2 * 2 + 256 * 4 * 3, stream>>>(
      pts, knn, W1h, lbs, cbias, t);
  cls_kernel<<<BATCH, 256, 0, stream>>>(cls_tok, t);

  const long tStride = (long)TOK * DMODEL;
  // transformer blocks
  for (int i = 0; i < 2; ++i) {
    ln_kernel<<<BATCH * TOK, 256, 0, stream>>>(t, ln1_g + i * 256, ln1_b + i * 256, y16);
    gemm_wmma_kernel<<<gemm_grid(TOK, 768, BATCH), 32, 0, stream>>>(
        y16, tStride, qkvW16 + (size_t)i * 768 * 256, nullptr, nullptr, nullptr, 0,
        qkvbuf, nullptr, (long)TOK * 768, TOK, 768, 256, 0);
    attn_kernel<<<BATCH * 4, 96, (TOK * 64 * 3 + TOK * TOK) * 4, stream>>>(qkvbuf, o16);
    gemm_wmma_kernel<<<gemm_grid(TOK, 256, BATCH), 32, 0, stream>>>(
        o16, tStride, projW16 + (size_t)i * 256 * 256, nullptr, proj_b + i * 256, t, tStride,
        t, nullptr, tStride, TOK, 256, 256, 0);
    ln_kernel<<<BATCH * TOK, 256, 0, stream>>>(t, ln2_g + i * 256, ln2_b + i * 256, y16);
    gemm_wmma_kernel<<<gemm_grid(TOK, 512, BATCH), 32, 0, stream>>>(
        y16, tStride, fc1W16 + (size_t)i * 512 * 256, nullptr, fc1_b + i * 512, nullptr, 0,
        nullptr, mlp16, (long)TOK * 512, TOK, 512, 256, 2);
    gemm_wmma_kernel<<<gemm_grid(TOK, 256, BATCH), 32, 0, stream>>>(
        mlp16, (long)TOK * 512, fc2W16 + (size_t)i * 256 * 512, nullptr, fc2_b + i * 256, t, tStride,
        t, nullptr, tStride, TOK, 256, 512, 0);
  }

  // head
  ln_kernel<<<BATCH * TOK, 256, 0, stream>>>(t, norm_g, norm_b, y16);
  gemm_wmma_kernel<<<gemm_grid(TOK, 1024, BATCH), 32, 0, stream>>>(
      y16, tStride, h0W16, h0s, h0h, nullptr, 0,
      h0out, nullptr, (long)TOK * 1024, TOK, 1024, 256, 1);
  pool_kernel<<<(BATCH * 1024) / 256, 256, 0, stream>>>(h0out, pooled16);
  gemm_wmma_kernel<<<gemm_grid(BATCH, 512, 1), 32, 0, stream>>>(
      pooled16, 0, h1W16, h1s, h1h, nullptr, 0,
      nullptr, z116, 0, BATCH, 512, 2048, 1);
  gemm_wmma_kernel<<<gemm_grid(BATCH, 64, 1), 32, 0, stream>>>(
      z116, 0, se1W16, ses, seh, nullptr, 0,
      nullptr, s116, 0, BATCH, 64, 512, 3);
  gemm_wmma_kernel<<<gemm_grid(BATCH, 512, 1), 32, 0, stream>>>(
      s116, 0, se2W16, nullptr, nullptr, nullptr, 0,
      s2buf, nullptr, 0, BATCH, 512, 64, 0);
  sescale_kernel<<<(BATCH * 512) / 256, 256, 0, stream>>>(z116, s2buf, z16);
  gemm_wmma_kernel<<<gemm_grid(BATCH, 40, 1), 32, 0, stream>>>(
      z16, 0, h2W16, nullptr, h2_b, nullptr, 0,
      (float*)d_out, nullptr, 0, BATCH, 40, 512, 0);
}
